// DotProductAttention_40106404610886
// MI455X (gfx1250) — matrix operand order repaired and checked by reading, not compiled, not run on hardware
//
#include <hip/hip_runtime.h>
#include <stdint.h>

#define DIM   64      // head dim (fixed by reference)
#define KB    64      // keys per inner tile (2 WMMA K-slices)
#define BQ    128     // queries per workgroup (8 waves x 16 rows)
#define NWAVE 8
#define KPAD  88      // LDS row pitch (halfwords) for K tile   (176B: 16B-aligned, 16 distinct banks)
#define VPAD  72      // LDS row pitch for V^T tile             (144B: 16B/8B-aligned, 16 distinct banks)
#define PPAD  72      // LDS row pitch for P transpose bounce

typedef __attribute__((ext_vector_type(16))) __bf16 v16bf;
typedef __attribute__((ext_vector_type(2)))  __bf16 v2bf;
typedef __attribute__((ext_vector_type(8)))  float  v8f;

#define LOG2E 1.44269504088896340736f

union BV { v16bf v; uint4 u[2]; };

// fp32 -> packed bf16x2, preferring the native converter
__device__ __forceinline__ unsigned pack2(float a, float b) {
#if __has_builtin(__builtin_amdgcn_cvt_pk_bf16_f32)
  v2bf t = __builtin_amdgcn_cvt_pk_bf16_f32(a, b);
  return __builtin_bit_cast(unsigned, t);
#else
  __bf16 ha = (__bf16)a, hb = (__bf16)b;
  unsigned short ua = __builtin_bit_cast(unsigned short, ha);
  unsigned short ub = __builtin_bit_cast(unsigned short, hb);
  return (unsigned)ua | ((unsigned)ub << 16);
#endif
}
__device__ __forceinline__ float fexp2(float x) {
#if __has_builtin(__builtin_amdgcn_exp2f)
  return __builtin_amdgcn_exp2f(x);
#else
  return exp2f(x);
#endif
}

__global__ __launch_bounds__(256) void fa_fwd_kernel(
    const float* __restrict__ Q, const float* __restrict__ K,
    const float* __restrict__ V, const int* __restrict__ VL,
    float* __restrict__ O, int S)
{
  __shared__ unsigned short Kl[2][KB][KPAD];    // K tile, bf16, row-major [local key][dim]
  __shared__ unsigned short Vt[2][DIM][VPAD];   // V tile, bf16, transposed [dim][physical key col]
  __shared__ unsigned short Pl[NWAVE][16][PPAD];// per-wave P bounce [row][physical key col]
  // physical key column c = 4*ln + t  <->  local key a = 16*t + ln  (P and V^T agree)

  const int b    = blockIdx.y;
  const int q0   = blockIdx.x * BQ;
  const int tid  = threadIdx.x;
  const int wave = tid >> 5;
  const int lane = tid & 31;
  const int hi   = lane >> 4;      // half-wave select
  const int ln   = lane & 15;
  const int k0   = hi * 8;         // A-operand K sub-offset per ISA layout

  const int valid = VL[b];
  const int kvend = min(S, (valid + KB - 1) & ~(KB - 1)); // masked tail tiles give exp2()==0

  const float* Qb = Q + (size_t)b * S * DIM;
  const float* Kg = K + (size_t)b * S * DIM;
  const float* Vg = V + (size_t)b * S * DIM;
  float*       Ob = O + (size_t)b * S * DIM;

  // ---- Q tile (16 rows / wave) -> two bf16 A operands, scaled by log2(e)/sqrt(D)
  BV qa0, qa1;
  {
    const float* qp = Qb + (size_t)(q0 + wave * 16 + ln) * DIM + k0;
    const float sc = 0.125f * LOG2E;      // scores produced directly in log2 domain
    float4 f0 = *(const float4*)(qp +  0), f1 = *(const float4*)(qp +  4);
    float4 f2 = *(const float4*)(qp + 16), f3 = *(const float4*)(qp + 20);
    float4 f4 = *(const float4*)(qp + 32), f5 = *(const float4*)(qp + 36);
    float4 f6 = *(const float4*)(qp + 48), f7 = *(const float4*)(qp + 52);
    qa0.u[0].x = pack2(f0.x*sc, f0.y*sc); qa0.u[0].y = pack2(f0.z*sc, f0.w*sc);
    qa0.u[0].z = pack2(f1.x*sc, f1.y*sc); qa0.u[0].w = pack2(f1.z*sc, f1.w*sc);
    qa0.u[1].x = pack2(f2.x*sc, f2.y*sc); qa0.u[1].y = pack2(f2.z*sc, f2.w*sc);
    qa0.u[1].z = pack2(f3.x*sc, f3.y*sc); qa0.u[1].w = pack2(f3.z*sc, f3.w*sc);
    qa1.u[0].x = pack2(f4.x*sc, f4.y*sc); qa1.u[0].y = pack2(f4.z*sc, f4.w*sc);
    qa1.u[0].z = pack2(f5.x*sc, f5.y*sc); qa1.u[0].w = pack2(f5.z*sc, f5.w*sc);
    qa1.u[1].x = pack2(f6.x*sc, f6.y*sc); qa1.u[1].y = pack2(f6.z*sc, f6.w*sc);
    qa1.u[1].z = pack2(f7.x*sc, f7.y*sc); qa1.u[1].w = pack2(f7.z*sc, f7.w*sc);
  }

  // all-ones bf16 B operand for row sums via the matrix pipe
  BV ones;
  ones.u[0].x = 0x3F803F80u; ones.u[0].y = 0x3F803F80u;
  ones.u[0].z = 0x3F803F80u; ones.u[0].w = 0x3F803F80u;
  ones.u[1] = ones.u[0];

  const v8f z = {0.f,0.f,0.f,0.f,0.f,0.f,0.f,0.f};
  v8f acc[4] = { z, z, z, z };         // O accumulator, 16x64 split into 4 N-tiles
  float mrow[8], lrow[8];              // flash stats for this half-wave's 8 rows
  #pragma unroll
  for (int r = 0; r < 8; ++r) { mrow[r] = -1e30f; lrow[r] = 0.f; }

  // ---- loader mappings
  const int kr  = tid >> 2;            // K: key row, 16 dims
  const int c0  = (tid & 3) << 4;
  const int vln = tid & 15;            // V: 4 stride-16 key rows x 4 dims
  const int vc4 = (tid >> 4) << 2;

  // prologue: stage tile 0
  float4 kA[4], vB[4];
  {
    const float* kp = Kg + (size_t)kr * DIM + c0;
    #pragma unroll
    for (int j = 0; j < 4; ++j) kA[j] = *(const float4*)(kp + j*4);
    #pragma unroll
    for (int t = 0; t < 4; ++t)
      vB[t] = *(const float4*)(Vg + (size_t)(16*t + vln) * DIM + vc4);
    uint4 w0, w1;
    w0.x = pack2(kA[0].x, kA[0].y); w0.y = pack2(kA[0].z, kA[0].w);
    w0.z = pack2(kA[1].x, kA[1].y); w0.w = pack2(kA[1].z, kA[1].w);
    w1.x = pack2(kA[2].x, kA[2].y); w1.y = pack2(kA[2].z, kA[2].w);
    w1.z = pack2(kA[3].x, kA[3].y); w1.w = pack2(kA[3].z, kA[3].w);
    *(uint4*)&Kl[0][kr][c0]     = w0;
    *(uint4*)&Kl[0][kr][c0 + 8] = w1;
    const float* b0 = (const float*)&vB[0];
    const float* b1 = (const float*)&vB[1];
    const float* b2 = (const float*)&vB[2];
    const float* b3 = (const float*)&vB[3];
    #pragma unroll
    for (int j = 0; j < 4; ++j) {
      uint2 w; w.x = pack2(b0[j], b1[j]); w.y = pack2(b2[j], b3[j]);
      *(uint2*)&Vt[0][vc4 + j][4 * vln] = w;
    }
  }
  __syncthreads();

  for (int kbase = 0; kbase < kvend; kbase += KB) {
    const int  cur     = (kbase >> 6) & 1;
    const bool hasNext = (kbase + KB) < kvend;

    if (hasNext) {   // issue next-tile global loads early (hidden behind WMMAs/softmax)
      const float* kp = Kg + (size_t)(kbase + KB + kr) * DIM + c0;
      #pragma unroll
      for (int j = 0; j < 4; ++j) kA[j] = *(const float4*)(kp + j*4);
      #pragma unroll
      for (int t = 0; t < 4; ++t)
        vB[t] = *(const float4*)(Vg + (size_t)(kbase + KB + 16*t + vln) * DIM + vc4);
      if (kbase + 2 * KB < kvend) {
        __builtin_prefetch(Kg + (size_t)(kbase + 2*KB + kr) * DIM + c0, 0, 1);
        __builtin_prefetch(Vg + (size_t)(kbase + 2*KB + vln) * DIM + vc4, 0, 1);
      }
    }

    // ---- scores: S = Q * K^T   (four 16-key N-tiles x two D-slices), log2 domain
    v8f sc4[4] = { z, z, z, z };
    {
      const unsigned short* Kc = &Kl[cur][0][0];
      #pragma unroll
      for (int t = 0; t < 4; ++t) {
        BV bk;
        const unsigned short* p = Kc + (size_t)(t * 16 + ln) * KPAD + hi * 8;
        bk.u[0] = *(const uint4*)(p     ); bk.u[1] = *(const uint4*)(p + 16);
        sc4[t] = __builtin_amdgcn_wmma_f32_16x16x32_bf16(false, qa0.v, false, bk.v, (short)0, sc4[t], false, false);
        bk.u[0] = *(const uint4*)(p + 32); bk.u[1] = *(const uint4*)(p + 48);
        sc4[t] = __builtin_amdgcn_wmma_f32_16x16x32_bf16(false, qa1.v, false, bk.v, (short)0, sc4[t], false, false);
      }
    }

    // ---- online softmax, max only (sum comes from the matrix pipe below)
    // P store: physical column c = 4*ln + t -> one b64 store per row
    unsigned short* Pw = &Pl[wave][0][0];
    float sclr[8];
    if (kbase + KB <= valid) {
      // steady state: no masking at all
      #pragma unroll
      for (int r = 0; r < 8; ++r) {
        float a0 = sc4[0][r], a1 = sc4[1][r], a2 = sc4[2][r], a3 = sc4[3][r];
        float mx = fmaxf(fmaxf(a0, a1), fmaxf(a2, a3));
        mx = fmaxf(mx, __shfl_xor(mx, 1, 16));
        mx = fmaxf(mx, __shfl_xor(mx, 2, 16));
        mx = fmaxf(mx, __shfl_xor(mx, 4, 16));
        mx = fmaxf(mx, __shfl_xor(mx, 8, 16));
        float mnew = fmaxf(mrow[r], mx);
        sclr[r] = fexp2(mrow[r] - mnew);
        mrow[r] = mnew;
        uint2 w;
        w.x = pack2(fexp2(a0 - mnew), fexp2(a1 - mnew));
        w.y = pack2(fexp2(a2 - mnew), fexp2(a3 - mnew));
        *(uint2*)&Pw[(r + hi * 8) * PPAD + 4 * ln] = w;
      }
    } else {
      // boundary tile: mask key columns >= valid (score (t,ln) <-> key kbase+16t+ln)
      const bool m0 = (kbase +      ln) < valid;
      const bool m1 = (kbase + 16 + ln) < valid;
      const bool m2 = (kbase + 32 + ln) < valid;
      const bool m3 = (kbase + 48 + ln) < valid;
      #pragma unroll
      for (int r = 0; r < 8; ++r) {
        float a0 = m0 ? sc4[0][r] : -1e6f;
        float a1 = m1 ? sc4[1][r] : -1e6f;
        float a2 = m2 ? sc4[2][r] : -1e6f;
        float a3 = m3 ? sc4[3][r] : -1e6f;
        float mx = fmaxf(fmaxf(a0, a1), fmaxf(a2, a3));
        mx = fmaxf(mx, __shfl_xor(mx, 1, 16));
        mx = fmaxf(mx, __shfl_xor(mx, 2, 16));
        mx = fmaxf(mx, __shfl_xor(mx, 4, 16));
        mx = fmaxf(mx, __shfl_xor(mx, 8, 16));
        float mnew = fmaxf(mrow[r], mx);
        sclr[r] = fexp2(mrow[r] - mnew);
        mrow[r] = mnew;
        uint2 w;
        w.x = pack2(fexp2(a0 - mnew), fexp2(a1 - mnew));
        w.y = pack2(fexp2(a2 - mnew), fexp2(a3 - mnew));
        *(uint2*)&Pw[(r + hi * 8) * PPAD + 4 * ln] = w;
      }
    }
    asm volatile("s_wait_dscnt 0x0" ::: "memory");   // intra-wave LDS RAW: C-layout -> A-layout

    // ---- P as A operands (two 32-physical-column slices)
    const unsigned short* pp = Pw + (size_t)ln * PPAD + k0;
    BV pa0, pa1;
    pa0.u[0] = *(const uint4*)(pp     );
    pa0.u[1] = *(const uint4*)(pp + 16);
    pa1.u[0] = *(const uint4*)(pp + 32);
    pa1.u[1] = *(const uint4*)(pp + 48);

    // ---- row sums on the matrix pipe: bs = P * ones (replicated across lanes in C layout)
    v8f bs = z;
    bs = __builtin_amdgcn_wmma_f32_16x16x32_bf16(false, pa0.v, false, ones.v, (short)0, bs, false, false);
    bs = __builtin_amdgcn_wmma_f32_16x16x32_bf16(false, pa1.v, false, ones.v, (short)0, bs, false, false);

    // ---- flash state update
    #pragma unroll
    for (int r = 0; r < 8; ++r) {
      lrow[r] = lrow[r] * sclr[r] + bs[r];
      acc[0][r] *= sclr[r]; acc[1][r] *= sclr[r];
      acc[2][r] *= sclr[r]; acc[3][r] *= sclr[r];
    }

    // ---- O += P * V  (two physical-column slices; V^T in LDS)
    {
      const unsigned short* Vc = &Vt[cur][0][0];
      #pragma unroll
      for (int t = 0; t < 4; ++t) {
        BV bv;
        const unsigned short* vp = Vc + (size_t)(t * 16 + ln) * VPAD + hi * 8;
        bv.u[0] = *(const uint4*)(vp    );
        bv.u[1] = *(const uint4*)(vp + 16);
        acc[t] = __builtin_amdgcn_wmma_f32_16x16x32_bf16(false, pa0.v, false, bv.v, (short)0, acc[t], false, false);
      }
      #pragma unroll
      for (int t = 0; t < 4; ++t) {
        BV bv;
        const unsigned short* vp = Vc + (size_t)(t * 16 + ln) * VPAD + 32 + hi * 8;
        bv.u[0] = *(const uint4*)(vp    );
        bv.u[1] = *(const uint4*)(vp + 16);
        acc[t] = __builtin_amdgcn_wmma_f32_16x16x32_bf16(false, pa1.v, false, bv.v, (short)0, acc[t], false, false);
      }
    }

    // ---- stage next tile into the other buffer
    if (hasNext) {
      const int nxt = cur ^ 1;
      uint4 w0, w1;
      w0.x = pack2(kA[0].x, kA[0].y); w0.y = pack2(kA[0].z, kA[0].w);
      w0.z = pack2(kA[1].x, kA[1].y); w0.w = pack2(kA[1].z, kA[1].w);
      w1.x = pack2(kA[2].x, kA[2].y); w1.y = pack2(kA[2].z, kA[2].w);
      w1.z = pack2(kA[3].x, kA[3].y); w1.w = pack2(kA[3].z, kA[3].w);
      *(uint4*)&Kl[nxt][kr][c0]     = w0;
      *(uint4*)&Kl[nxt][kr][c0 + 8] = w1;
      const float* b0 = (const float*)&vB[0];
      const float* b1 = (const float*)&vB[1];
      const float* b2 = (const float*)&vB[2];
      const float* b3 = (const float*)&vB[3];
      #pragma unroll
      for (int j = 0; j < 4; ++j) {
        uint2 w; w.x = pack2(b0[j], b1[j]); w.y = pack2(b2[j], b3[j]);
        *(uint2*)&Vt[nxt][vc4 + j][4 * vln] = w;
      }
    }
    __syncthreads();
  }

  // ---- normalize and store (fp32 out)
  #pragma unroll
  for (int r = 0; r < 8; ++r) {
    const float inv = 1.0f / lrow[r];
    float* op = Ob + (size_t)(q0 + wave * 16 + r + hi * 8) * DIM + ln;
    op[ 0] = acc[0][r] * inv;
    op[16] = acc[1][r] * inv;
    op[32] = acc[2][r] * inv;
    op[48] = acc[3][r] * inv;
  }
}

extern "C" void kernel_launch(void* const* d_in, const int* in_sizes, int n_in,
                              void* d_out, int out_size, void* d_ws, size_t ws_size,
                              hipStream_t stream) {
  const float* q  = (const float*)d_in[0];
  const float* k  = (const float*)d_in[1];
  const float* v  = (const float*)d_in[2];
  const int*   vl = (const int*)d_in[3];
  float* out = (float*)d_out;

  const int B = in_sizes[3];
  const int S = in_sizes[0] / (B * DIM);

  dim3 grid(S / BQ, B);
  fa_fwd_kernel<<<grid, 256, 0, stream>>>(q, k, v, vl, out, S);
  (void)d_ws; (void)ws_size; (void)n_in; (void)out_size;
}
